// RPN_31937376813767
// MI455X (gfx1250) — matrix operand order repaired and checked by reading, not compiled, not run on hardware
//
#include <hip/hip_runtime.h>
#include <hip/hip_bf16.h>
#include <math.h>

typedef __attribute__((ext_vector_type(16))) _Float16 v16h;
typedef __attribute__((ext_vector_type(8)))  _Float16 v8h;
typedef __attribute__((ext_vector_type(8)))  float    v8f;

#define BATCH 4
#define CH    256
#define Hf    128
#define Wf    128
#define HWf   (Hf*Wf)        /* 16384 */
#define Pf    130            /* padded edge */
#define NANCH 9
#define NBOX  (HWf*NANCH)    /* 147456 */
#define NHEAD 48
#define MAXP  1000

/* ---- d_out float offsets (props, scs, valids, objectness, deltas) ---- */
#define OUT_PROPS 0
#define OUT_SCS   16000
#define OUT_VALID 20000
#define OUT_OBJ   24000
#define OUT_DEL   613824      /* 24000 + 4*147456 */

/* ---- workspace byte offsets ---- */
#define WS_XPAD   0ull          /* padded NHWC f16 features: 4*130*130*256*2 */
#define WS_X1H    34611200ull   /* NHWC f16 conv out: 32 MB */
#define WS_WPACK  68165632ull   /* packed 3x3 weights: 2304*256 f16 */
#define WS_HWPACK 69345280ull   /* packed head weights: 256*48 f16 */
#define WS_BOXES  69369856ull   /* B*NBOX*4 f32 */
#define WS_SCORES 78807040ull   /* B*NBOX f32 */
#define WS_S0     81166336ull   /* B*NBOX f32 */

union AFrag { v16h v; v8h h[2]; };

/* ---------------- K0: NCHW f32 -> zero-halo-padded NHWC f16 ---------------- */
__global__ void k_pad_nhwc_h(const float* __restrict__ in, _Float16* __restrict__ out) {
    size_t tid = (size_t)blockIdx.x * blockDim.x + threadIdx.x;
    size_t total = (size_t)BATCH * Pf * Pf * CH;
    if (tid >= total) return;
    int c  = (int)(tid % CH);
    size_t t = tid / CH;
    int xx = (int)(t % Pf);  t /= Pf;
    int yy = (int)(t % Pf);
    int b  = (int)(t / Pf);
    _Float16 v = (_Float16)0.f;
    if (xx >= 1 && xx <= Wf && yy >= 1 && yy <= Hf)
        v = (_Float16)in[((size_t)b * CH + c) * HWf + (size_t)(yy - 1) * Wf + (xx - 1)];
    out[tid] = v;
}

/* ---------------- K1: pack 3x3 conv weights into WMMA B-fragment order ----------------
 * Logical B[k][n]: k = tap*256 + cin (tap = dy*3+dx), n = cout.
 * 16-bit B 32x16 fragment: lane<16 -> K 0..7,16..23 of chunk, lane>=16 -> K 8..15,24..31,
 * N = lane&15, 16 halfs per lane. Stored wpack[(tile*32+lane)*16 + j]. */
__global__ void k_pack_w3(const float* __restrict__ cw, _Float16* __restrict__ wpack) {
    int tid = blockIdx.x * blockDim.x + threadIdx.x;           /* 72*16*32 = 36864 */
    if (tid >= 72 * 16 * 32) return;
    int lane = tid & 31;
    int tile = tid >> 5;
    int kt = tile >> 4;        /* 0..71 : K chunk of 32 */
    int nt = tile & 15;        /* 0..15 : N tile */
    int koff = (lane >= 16) ? 8 : 0;
    int n = nt * 16 + (lane & 15);                             /* cout */
    #pragma unroll
    for (int j = 0; j < 16; ++j) {
        int kk  = kt * 32 + koff + j + ((j >= 8) ? 8 : 0);     /* global K */
        int tap = kk >> 8;
        int cin = kk & 255;
        float v = cw[(size_t)n * (CH * 9) + (size_t)cin * 9 + tap];
        wpack[((size_t)tile * 32 + lane) * 16 + j] = (_Float16)v;
    }
}

/* ---------------- K2: pack head (cls 9 + reg 36, padded to 48) weights ---------------- */
__global__ void k_pack_whead(const float* __restrict__ clsw, const float* __restrict__ regw,
                             _Float16* __restrict__ hw) {
    int tid = blockIdx.x * blockDim.x + threadIdx.x;           /* 8*3*32 = 768 */
    if (tid >= 8 * 3 * 32) return;
    int lane = tid & 31;
    int tile = tid >> 5;
    int kt = tile / 3;
    int nt = tile % 3;
    int koff = (lane >= 16) ? 8 : 0;
    int n = nt * 16 + (lane & 15);                             /* 0..47 */
    #pragma unroll
    for (int j = 0; j < 16; ++j) {
        int cin = kt * 32 + koff + j + ((j >= 8) ? 8 : 0);
        float v = 0.f;
        if (n < 9)       v = clsw[(size_t)n * CH + cin];
        else if (n < 45) v = regw[(size_t)(n - 9) * CH + cin];
        hw[((size_t)tile * 32 + lane) * 16 + j] = (_Float16)v;
    }
}

/* ---------------- K3: 3x3 conv + bias + ReLU as implicit GEMM on WMMA ----------------
 * One wave computes 16(M=spatial) x 64(N=cout): 4 accumulators sharing one A fragment.
 * Halo-padded input -> no divergence; K loop = 9 taps x 8 chunks of 32. */
__global__ __launch_bounds__(256)
void k_conv3x3_wmma(const _Float16* __restrict__ xpad, const _Float16* __restrict__ wpack,
                    const float* __restrict__ cb, _Float16* __restrict__ x1h) {
    int lane  = threadIdx.x & 31;
    int wgid  = blockIdx.x * 8 + (threadIdx.x >> 5);
    int mtile = wgid >> 2;            /* 0..16383 */
    int ntg   = wgid & 3;             /* N group of 64 couts */
    int b      = mtile >> 10;
    int within = mtile & 1023;        /* y*8 + xtile */
    int y  = within >> 3;
    int x0 = (within & 7) << 4;
    int m  = lane & 15;
    int sel0 = (lane < 16) ? 0 : 8;   /* A fragment K-half selector */

    v8f acc[4] = {};
    for (int tap = 0; tap < 9; ++tap) {
        int dy = tap / 3;
        int dx = tap % 3;
        const _Float16* abase = xpad +
            ((size_t)b * (Pf * Pf) + (size_t)(y + dy) * Pf + (x0 + m + dx)) * CH;
        if (tap < 8) {
            int dy2 = (tap + 1) / 3, dx2 = (tap + 1) % 3;
            __builtin_prefetch(xpad + ((size_t)b * (Pf * Pf) + (size_t)(y + dy2) * Pf +
                                       (x0 + m + dx2)) * CH, 0, 0);
        }
        #pragma unroll
        for (int cc = 0; cc < 8; ++cc) {
            AFrag a;
            a.h[0] = *(const v8h*)(abase + cc * 32 + sel0);
            a.h[1] = *(const v8h*)(abase + cc * 32 + sel0 + 16);
            const _Float16* wb = wpack +
                (((size_t)((tap * 8 + cc) * 16 + ntg * 4) * 32 + lane) << 4);
            v16h b0 = *(const v16h*)(wb);
            v16h b1 = *(const v16h*)(wb + 512);
            v16h b2 = *(const v16h*)(wb + 1024);
            v16h b3 = *(const v16h*)(wb + 1536);
            acc[0] = __builtin_amdgcn_wmma_f32_16x16x32_f16(false, a.v, false, b0, (short)0, acc[0], false, false);
            acc[1] = __builtin_amdgcn_wmma_f32_16x16x32_f16(false, a.v, false, b1, (short)0, acc[1], false, false);
            acc[2] = __builtin_amdgcn_wmma_f32_16x16x32_f16(false, a.v, false, b2, (short)0, acc[2], false, false);
            acc[3] = __builtin_amdgcn_wmma_f32_16x16x32_f16(false, a.v, false, b3, (short)0, acc[3], false, false);
        }
    }
    int mbase = (lane < 16) ? 0 : 8;
    size_t rowbase = (size_t)b * HWf + (size_t)y * Wf + x0;
    #pragma unroll
    for (int q = 0; q < 4; ++q) {
        int n = (ntg * 4 + q) * 16 + (lane & 15);
        float bias = cb[n];
        #pragma unroll
        for (int v = 0; v < 8; ++v) {
            float val = acc[q][v] + bias;
            val = val > 0.f ? val : 0.f;
            x1h[(rowbase + v + mbase) * CH + n] = (_Float16)val;
        }
    }
}

/* ---------------- K4: head 1x1 convs (cls + reg fused, N=48) on WMMA ----------------
 * One wave computes 16(M) x 48(N): 3 accumulators sharing one A fragment.
 * Writes objectness and bbox_deltas directly into d_out (NHWC flatten order). */
__global__ __launch_bounds__(256)
void k_head_wmma(const _Float16* __restrict__ x1h, const _Float16* __restrict__ hw,
                 const float* __restrict__ clsb, const float* __restrict__ regb,
                 float* __restrict__ out) {
    int lane  = threadIdx.x & 31;
    int wgid  = blockIdx.x * 8 + (threadIdx.x >> 5);   /* 4096 waves */
    int g0 = wgid * 16;                                /* global row over B*HW */
    int m  = lane & 15;
    int sel0 = (lane < 16) ? 0 : 8;

    v8f acc[3] = {};
    const _Float16* abase = x1h + (size_t)(g0 + m) * CH;
    #pragma unroll
    for (int cc = 0; cc < 8; ++cc) {
        AFrag a;
        a.h[0] = *(const v8h*)(abase + cc * 32 + sel0);
        a.h[1] = *(const v8h*)(abase + cc * 32 + sel0 + 16);
        const _Float16* wb = hw + (((size_t)(cc * 3) * 32 + lane) << 4);
        v16h b0 = *(const v16h*)(wb);
        v16h b1 = *(const v16h*)(wb + 512);
        v16h b2 = *(const v16h*)(wb + 1024);
        acc[0] = __builtin_amdgcn_wmma_f32_16x16x32_f16(false, a.v, false, b0, (short)0, acc[0], false, false);
        acc[1] = __builtin_amdgcn_wmma_f32_16x16x32_f16(false, a.v, false, b1, (short)0, acc[1], false, false);
        acc[2] = __builtin_amdgcn_wmma_f32_16x16x32_f16(false, a.v, false, b2, (short)0, acc[2], false, false);
    }
    int mbase = (lane < 16) ? 0 : 8;
    #pragma unroll
    for (int nt = 0; nt < 3; ++nt) {
        int n = nt * 16 + (lane & 15);                 /* 0..47 */
        float bias = (n < 9) ? clsb[n] : ((n < 45) ? regb[n - 9] : 0.f);
        #pragma unroll
        for (int v = 0; v < 8; ++v) {
            int g = g0 + v + mbase;
            int b = g >> 14;          /* /HWf */
            int p = g & (HWf - 1);
            float val = acc[nt][v] + bias;
            if (n < 9)
                out[OUT_OBJ + ((size_t)b * HWf + p) * 9 + n] = val;
            else if (n < 45)
                out[OUT_DEL + ((size_t)b * HWf + p) * 36 + (n - 9)] = val;
        }
    }
}

/* ---------------- K5: sigmoid + anchor decode + clip + score threshold ---------------- */
__global__ void k_decode(const float* __restrict__ out, float* __restrict__ boxes,
                         float* __restrict__ scores, float* __restrict__ s0) {
    int idx = blockIdx.x * blockDim.x + threadIdx.x;   /* B*NBOX */
    if (idx >= BATCH * NBOX) return;
    int j = idx % NBOX;
    int p = j / NANCH;
    int a = j % NANCH;
    int y = p >> 7;
    int x = p & 127;
    const float SC[3] = {0.5f, 1.0f, 2.0f};
    float size = 32.f * SC[a / 3];
    float sr = sqrtf(SC[a % 3]);
    float ah = size * sr;
    float aw = size / sr;
    float cx = (float)x * 8.f;
    float cy = (float)y * 8.f;

    float obj = out[OUT_OBJ + idx];
    float dxv = out[OUT_DEL + (size_t)idx * 4 + 0];
    float dyv = out[OUT_DEL + (size_t)idx * 4 + 1];
    float dwv = out[OUT_DEL + (size_t)idx * 4 + 2];
    float dhv = out[OUT_DEL + (size_t)idx * 4 + 3];

    float score = 1.f / (1.f + expf(-obj));
    float pcx = dxv * aw + cx;
    float pcy = dyv * ah + cy;
    float pw  = expf(dwv) * aw;
    float ph  = expf(dhv) * ah;
    float x1 = fminf(fmaxf(pcx - 0.5f * pw, 0.f), 1024.f);
    float y1 = fminf(fmaxf(pcy - 0.5f * ph, 0.f), 1024.f);
    float x2 = fminf(fmaxf(pcx + 0.5f * pw, 0.f), 1024.f);
    float y2 = fminf(fmaxf(pcy + 0.5f * ph, 0.f), 1024.f);

    boxes[(size_t)idx * 4 + 0] = x1;
    boxes[(size_t)idx * 4 + 1] = y1;
    boxes[(size_t)idx * 4 + 2] = x2;
    boxes[(size_t)idx * 4 + 3] = y2;
    scores[idx] = score;
    s0[idx] = (score >= 0.5f) ? score : -1.0f;
}

/* ---------------- K6: sequential greedy NMS, one block (1024 thr) per image ---------------- */
__global__ __launch_bounds__(1024)
void k_nms(const float* __restrict__ boxes_all, const float* __restrict__ scores_all,
           float* __restrict__ s0_all, float* __restrict__ out) {
    __shared__ float sv[1024];
    __shared__ int   si[1024];
    __shared__ float cb[4];
    __shared__ float cscore;
    int b = blockIdx.x;
    int tid = threadIdx.x;
    const float* boxes  = boxes_all  + (size_t)b * NBOX * 4;
    const float* scores = scores_all + (size_t)b * NBOX;
    float* s0 = s0_all + (size_t)b * NBOX;

    for (int it = 0; it < MAXP; ++it) {
        /* block argmax (first-index tiebreak to match jnp.argmax) */
        float bestv = -1e30f;
        int   besti = 0;
        for (int k = tid; k < NBOX; k += 1024) {
            float v = s0[k];
            if (v > bestv) { bestv = v; besti = k; }
        }
        sv[tid] = bestv; si[tid] = besti;
        __syncthreads();
        for (int s = 512; s > 0; s >>= 1) {
            if (tid < s) {
                float v2 = sv[tid + s]; int i2 = si[tid + s];
                if (v2 > sv[tid] || (v2 == sv[tid] && i2 < si[tid])) {
                    sv[tid] = v2; si[tid] = i2;
                }
            }
            __syncthreads();
        }
        int   idx   = si[0];
        float val   = sv[0];
        bool  valid = val > 0.0f;
        if (tid == 0) {
            cb[0] = boxes[(size_t)idx * 4 + 0];
            cb[1] = boxes[(size_t)idx * 4 + 1];
            cb[2] = boxes[(size_t)idx * 4 + 2];
            cb[3] = boxes[(size_t)idx * 4 + 3];
            cscore = scores[idx];
        }
        __syncthreads();
        if (tid == 0) {
            float vf = valid ? 1.f : 0.f;
            size_t o = (size_t)b * MAXP + it;
            out[OUT_PROPS + o * 4 + 0] = cb[0] * vf;
            out[OUT_PROPS + o * 4 + 1] = cb[1] * vf;
            out[OUT_PROPS + o * 4 + 2] = cb[2] * vf;
            out[OUT_PROPS + o * 4 + 3] = cb[3] * vf;
            out[OUT_SCS   + o] = cscore * vf;
            out[OUT_VALID + o] = vf;
        }
        if (valid) {
            float bx1 = cb[0], by1 = cb[1], bx2 = cb[2], by2 = cb[3];
            float barea = (bx2 - bx1) * (by2 - by1);
            for (int k = tid; k < NBOX; k += 1024) {
                const float* bp = boxes + (size_t)k * 4;
                float ix1 = fmaxf(bx1, bp[0]);
                float iy1 = fmaxf(by1, bp[1]);
                float ix2 = fminf(bx2, bp[2]);
                float iy2 = fminf(by2, bp[3]);
                float inter = fmaxf(ix2 - ix1, 0.f) * fmaxf(iy2 - iy1, 0.f);
                float area  = (bp[2] - bp[0]) * (bp[3] - bp[1]);
                float iou = inter / fmaxf(barea + area - inter, 1e-9f);
                if (iou > 0.7f) s0[k] = -1.0f;
            }
        }
        __threadfence_block();
        __syncthreads();
    }
}

extern "C" void kernel_launch(void* const* d_in, const int* in_sizes, int n_in,
                              void* d_out, int out_size, void* d_ws, size_t ws_size,
                              hipStream_t stream) {
    const float* features = (const float*)d_in[0];
    const float* conv_w   = (const float*)d_in[1];
    const float* conv_b   = (const float*)d_in[2];
    const float* cls_w    = (const float*)d_in[3];
    const float* cls_b    = (const float*)d_in[4];
    const float* reg_w    = (const float*)d_in[5];
    const float* reg_b    = (const float*)d_in[6];
    (void)in_sizes; (void)n_in; (void)out_size; (void)ws_size;

    char* ws = (char*)d_ws;
    _Float16* xpad   = (_Float16*)(ws + WS_XPAD);
    _Float16* x1h    = (_Float16*)(ws + WS_X1H);
    _Float16* wpack  = (_Float16*)(ws + WS_WPACK);
    _Float16* hwpack = (_Float16*)(ws + WS_HWPACK);
    float*    boxes  = (float*)(ws + WS_BOXES);
    float*    scores = (float*)(ws + WS_SCORES);
    float*    s0     = (float*)(ws + WS_S0);
    float*    out    = (float*)d_out;

    /* K0: features NCHW f32 -> zero-padded NHWC f16 */
    {
        size_t total = (size_t)BATCH * Pf * Pf * CH;
        k_pad_nhwc_h<<<(unsigned)((total + 255) / 256), 256, 0, stream>>>(features, xpad);
    }
    /* K1/K2: weight packing into WMMA fragment order */
    k_pack_w3<<<(72 * 16 * 32 + 255) / 256, 256, 0, stream>>>(conv_w, wpack);
    k_pack_whead<<<(8 * 3 * 32 + 255) / 256, 256, 0, stream>>>(cls_w, reg_w, hwpack);
    /* K3: 3x3 conv + ReLU (implicit GEMM, WMMA): 16384 M-tiles x 4 N-groups, 8 waves/block */
    k_conv3x3_wmma<<<(16384 * 4) / 8, 256, 0, stream>>>(xpad, wpack, conv_b, x1h);
    /* K4: fused cls/reg heads (WMMA): 4096 waves, writes objectness + deltas into d_out */
    k_head_wmma<<<4096 / 8, 256, 0, stream>>>(x1h, hwpack, cls_b, reg_b, out);
    /* K5: sigmoid + decode + clip + threshold */
    k_decode<<<(BATCH * NBOX + 255) / 256, 256, 0, stream>>>(out, boxes, scores, s0);
    /* K6: greedy NMS, one block per image */
    k_nms<<<BATCH, 1024, 0, stream>>>(boxes, scores, s0, out);
}
